// EncoderTreeRNN_4947802325448
// MI455X (gfx1250) — matrix operand order repaired and checked by reading, not compiled, not run on hardware
//
#include <hip/hip_runtime.h>
#include <hip/hip_bf16.h>

// ---------------------------------------------------------------------------
// Types for WMMA fragments (trivial ext-vector types so unions are legal)
// ---------------------------------------------------------------------------
typedef __bf16        v16bf  __attribute__((ext_vector_type(16)));
typedef float         v8f    __attribute__((ext_vector_type(8)));
typedef unsigned int  uint4v __attribute__((ext_vector_type(4)));
typedef float         flt4v  __attribute__((ext_vector_type(4)));

union Frag { v16bf v; uint4v q[2]; };

struct Ptrs9 { const float* p[9]; };

// weight order matches setup_inputs(): Wz Uzl Uzr Wr Url Urr Wh Uhl Uhr
enum { WZ = 0, UZL = 1, UZR = 2, WR = 3, URL = 4, URR = 5, WH = 6, UHL = 7, UHR = 8 };

#define MATW 16384  // 128*128 packed bf16 elements per matrix

__device__ __forceinline__ unsigned short f2bf(float f) {
    unsigned int u = __float_as_uint(f);
    unsigned int r = u + 0x7FFFu + ((u >> 16) & 1u);  // round-to-nearest-even
    return (unsigned short)(r >> 16);
}
__device__ __forceinline__ float bf2f(unsigned short h) {
    return __uint_as_float(((unsigned int)h) << 16);
}
__device__ __forceinline__ float sigm(float x) { return 1.0f / (1.0f + __expf(-x)); }

__device__ __forceinline__ v8f vzero8() {
    v8f v = {0.f, 0.f, 0.f, 0.f, 0.f, 0.f, 0.f, 0.f};
    return v;
}

// Load one 16x32 bf16 A-fragment from a row-major 64x128 bf16 LDS tile,
// following the documented 16-bit A-matrix layout:
//   lanes 0-15 : row m0+lane,    K = kb*32 + {0..7, 16..23}
//   lanes 16-31: row m0+lane-16, K = kb*32 + {8..15, 24..31}
__device__ __forceinline__ Frag load_a_frag(const unsigned short (*sm)[128],
                                            int m0, int lane, int kb) {
    Frag a;
    int row = m0 + (lane & 15);
    int k0  = kb * 32 + ((lane < 16) ? 0 : 8);
    a.q[0] = *(const uint4v*)&sm[row][k0];
    a.q[1] = *(const uint4v*)&sm[row][k0 + 16];
    return a;
}

// Load one 32x16 bf16 B-fragment from the packed weight buffer:
// packed layout = [ntile][kblock][lane][16 contiguous bf16] -> 2x b128, coalesced.
__device__ __forceinline__ Frag load_b_frag(const unsigned short* w,
                                            int nt, int kb, int lane) {
    Frag b;
    const uint4v* p = (const uint4v*)(w + (((nt * 4 + kb) * 32 + lane) << 4));
    b.q[0] = p[0];
    b.q[1] = p[1];
    return b;
}

#define WMMA_BF16(A, B, C) \
    __builtin_amdgcn_wmma_f32_16x16x32_bf16(false, (A), false, (B), (short)0, (C), false, false)

// ---------------------------------------------------------------------------
// Kernel 1: repack 9 weight matrices (128x128 f32, row-major [K][N]) into
// bf16 B-fragment order, and fold the 9 biases into 3 combined vectors.
// ---------------------------------------------------------------------------
__global__ __launch_bounds__(256) void tree_gru_pack(Ptrs9 w, Ptrs9 b,
                                                     unsigned short* pw,
                                                     float* bias) {
    long long idx = (long long)blockIdx.x * 256 + threadIdx.x;
    if (idx < 9LL * MATW) {
        int m    = (int)(idx / MATW);
        int e    = (int)(idx % MATW);
        int j    = e & 15;
        int lane = (e >> 4) & 31;
        int kb   = (e >> 9) & 3;
        int nt   = (e >> 11) & 7;
        // B-matrix striping: lanes 0-15 hold K = kb*32 + {0..7, 16..23}, lanes 16-31 hold +8
        int K = kb * 32 + ((lane < 16) ? 0 : 8) + (j & 7) + ((j >> 3) << 4);
        int N = nt * 16 + (lane & 15);
        pw[idx] = f2bf(w.p[m][K * 128 + N]);
    }
    if (blockIdx.x == 0 && threadIdx.x < 128) {
        int n = threadIdx.x;
        bias[n]       = b.p[WZ][n] + b.p[UZL][n] + b.p[UZR][n];  // bz
        bias[128 + n] = b.p[WR][n] + b.p[URL][n] + b.p[URR][n];  // br
        bias[256 + n] = b.p[WH][n] + b.p[UHL][n] + b.p[UHR][n];  // bh
    }
}

// ---------------------------------------------------------------------------
// Kernel 2: level 0.  h = (1-z)*tanh(x@Wh + bh), z = sigmoid(x@Wz + bz),
// x = emb[token].  64 rows x 128 cols per block (4 waves, 16-row strips).
// ---------------------------------------------------------------------------
__global__ __launch_bounds__(128) void tree_gru_level0(const int* __restrict__ tokens,
                                                       const float* __restrict__ emb,
                                                       const unsigned short* __restrict__ pw,
                                                       const float* __restrict__ bias,
                                                       unsigned short* __restrict__ out) {
    __shared__ __align__(16) unsigned short smA[64][128];

    const int tid  = threadIdx.x;
    const int lane = tid & 31;
    const int wave = tid >> 5;
    const long long rowBase = (long long)blockIdx.x * 64;

    // gather embeddings, convert to bf16 into LDS (coalesced float4 reads)
    for (int i = tid; i < 64 * 32; i += 128) {
        int r  = i >> 5;
        int c4 = (i & 31) << 2;
        int tok = tokens[rowBase + r];
        flt4v f = *(const flt4v*)(emb + (long long)tok * 128 + c4);
        smA[r][c4 + 0] = f2bf(f[0]);
        smA[r][c4 + 1] = f2bf(f[1]);
        smA[r][c4 + 2] = f2bf(f[2]);
        smA[r][c4 + 3] = f2bf(f[3]);
    }
    __syncthreads();

    const int m0 = wave * 16;
    v8f accz[8], acch[8];
#pragma unroll
    for (int nt = 0; nt < 8; nt++) { accz[nt] = vzero8(); acch[nt] = vzero8(); }

    const unsigned short* Wz = pw + WZ * MATW;
    const unsigned short* Wh = pw + WH * MATW;
#pragma unroll
    for (int kb = 0; kb < 4; kb++) {
        Frag a = load_a_frag(smA, m0, lane, kb);
#pragma unroll
        for (int nt = 0; nt < 8; nt++) {
            Frag bz = load_b_frag(Wz, nt, kb, lane);
            Frag bh = load_b_frag(Wh, nt, kb, lane);
            accz[nt] = WMMA_BF16(a.v, bz.v, accz[nt]);
            acch[nt] = WMMA_BF16(a.v, bh.v, acch[nt]);
        }
    }

    // C/D layout: lanes 0-15 -> n=lane, m=m0+g ; lanes 16-31 -> n=lane-16, m=m0+8+g
    const int nloc = lane & 15;
    const int mOff = (lane < 16) ? 0 : 8;
#pragma unroll
    for (int nt = 0; nt < 8; nt++) {
        int col = nt * 16 + nloc;
        float bzv = bias[col];
        float bhv = bias[256 + col];
#pragma unroll
        for (int g = 0; g < 8; g++) {
            float z  = sigm(accz[nt][g] + bzv);
            float ht = tanhf(acch[nt][g] + bhv);
            long long m = rowBase + m0 + mOff + g;
            out[m * 128 + col] = f2bf((1.0f - z) * ht);
        }
    }
}

// ---------------------------------------------------------------------------
// Kernel 3: one tree-reduction level.  cur: (Bb, 2*n2, 128) bf16 ->
// out: (Bb, n2, 128).  64 parents per block (4 waves, 16-parent strips).
// ---------------------------------------------------------------------------
__global__ __launch_bounds__(128) void tree_gru_reduce(const unsigned short* __restrict__ cur,
                                                       const unsigned short* __restrict__ pw,
                                                       const float* __restrict__ bias,
                                                       unsigned short* __restrict__ out_bf,
                                                       float* __restrict__ out_f32,
                                                       int n2, int final_level) {
    __shared__ __align__(16) unsigned short smHL[64][128];
    __shared__ __align__(16) unsigned short smHR[64][128];
    __shared__ __align__(16) unsigned short smRL[64][128];
    __shared__ __align__(16) unsigned short smRR[64][128];

    const int tid  = threadIdx.x;
    const int lane = tid & 31;
    const int wave = tid >> 5;
    const long long p0 = (long long)blockIdx.x * 64;

    // load left/right children (bf16) into LDS, dword granularity, coalesced
    for (int i = tid; i < 64 * 64; i += 128) {
        int r  = i >> 6;
        int c2 = i & 63;
        long long p = p0 + r;
        long long bidx = p / n2;
        long long j    = p - bidx * n2;
        long long lrow = (bidx * n2 + j) * 2;  // left child row; right = +1
        const unsigned int* ls = (const unsigned int*)(cur + lrow * 128);
        *(unsigned int*)&smHL[r][c2 * 2] = ls[c2];
        *(unsigned int*)&smHR[r][c2 * 2] = ls[64 + c2];
    }
    __syncthreads();

    const int m0 = wave * 16;
    const unsigned short* Uzl = pw + UZL * MATW;
    const unsigned short* Uzr = pw + UZR * MATW;
    const unsigned short* Url = pw + URL * MATW;
    const unsigned short* Urr = pw + URR * MATW;
    const unsigned short* Uhl = pw + UHL * MATW;
    const unsigned short* Uhr = pw + UHR * MATW;

    // ---- stage 1: z and r ----
    v8f accz[8], accr[8];
#pragma unroll
    for (int nt = 0; nt < 8; nt++) { accz[nt] = vzero8(); accr[nt] = vzero8(); }

#pragma unroll
    for (int kb = 0; kb < 4; kb++) {
        Frag al = load_a_frag(smHL, m0, lane, kb);
        Frag ar = load_a_frag(smHR, m0, lane, kb);
#pragma unroll
        for (int nt = 0; nt < 8; nt++) {
            Frag bzl = load_b_frag(Uzl, nt, kb, lane);
            Frag bzr = load_b_frag(Uzr, nt, kb, lane);
            Frag brl = load_b_frag(Url, nt, kb, lane);
            Frag brr = load_b_frag(Urr, nt, kb, lane);
            accz[nt] = WMMA_BF16(al.v, bzl.v, accz[nt]);
            accz[nt] = WMMA_BF16(ar.v, bzr.v, accz[nt]);
            accr[nt] = WMMA_BF16(al.v, brl.v, accr[nt]);
            accr[nt] = WMMA_BF16(ar.v, brr.v, accr[nt]);
        }
    }

    // elementwise: z kept in accz; r*hl, r*hr written back to LDS (bf16).
    // Each wave only touches its own 16 rows; same-wave DS ops are in-order.
    const int nloc = lane & 15;
    const int mOff = (lane < 16) ? 0 : 8;
#pragma unroll
    for (int nt = 0; nt < 8; nt++) {
        int col = nt * 16 + nloc;
        float bzv = bias[col];
        float brv = bias[128 + col];
#pragma unroll
        for (int g = 0; g < 8; g++) {
            int mt = m0 + mOff + g;
            float z = sigm(accz[nt][g] + bzv);
            float r = sigm(accr[nt][g] + brv);
            accz[nt][g] = z;
            smRL[mt][col] = f2bf(r * bf2f(smHL[mt][col]));
            smRR[mt][col] = f2bf(r * bf2f(smHR[mt][col]));
        }
    }

    // ---- stage 2: h_tilde ----
    v8f acch[8];
#pragma unroll
    for (int nt = 0; nt < 8; nt++) acch[nt] = vzero8();

#pragma unroll
    for (int kb = 0; kb < 4; kb++) {
        Frag al = load_a_frag(smRL, m0, lane, kb);
        Frag ar = load_a_frag(smRR, m0, lane, kb);
#pragma unroll
        for (int nt = 0; nt < 8; nt++) {
            Frag bhl = load_b_frag(Uhl, nt, kb, lane);
            Frag bhr = load_b_frag(Uhr, nt, kb, lane);
            acch[nt] = WMMA_BF16(al.v, bhl.v, acch[nt]);
            acch[nt] = WMMA_BF16(ar.v, bhr.v, acch[nt]);
        }
    }

    // ---- combine: out = z*(hl+hr) + (1-z)*tanh(acch + bh) ----
#pragma unroll
    for (int nt = 0; nt < 8; nt++) {
        int col = nt * 16 + nloc;
        float bhv = bias[256 + col];
#pragma unroll
        for (int g = 0; g < 8; g++) {
            int mt = m0 + mOff + g;
            float z  = accz[nt][g];
            float hl = bf2f(smHL[mt][col]);
            float hr = bf2f(smHR[mt][col]);
            float ht = tanhf(acch[nt][g] + bhv);
            float o  = z * (hl + hr) + (1.0f - z) * ht;
            long long prow = p0 + mt;
            if (final_level) out_f32[prow * 128 + col] = o;
            else             out_bf[prow * 128 + col]  = f2bf(o);
        }
    }
}

// ---------------------------------------------------------------------------
// Host launcher
// ---------------------------------------------------------------------------
extern "C" void kernel_launch(void* const* d_in, const int* in_sizes, int n_in,
                              void* d_out, int out_size, void* d_ws, size_t ws_size,
                              hipStream_t stream) {
    (void)in_sizes; (void)n_in; (void)out_size; (void)ws_size;

    const int*   tokens = (const int*)d_in[0];
    const float* emb    = (const float*)d_in[1];
    Ptrs9 w, b;
    for (int i = 0; i < 9; i++) {
        w.p[i] = (const float*)d_in[2 + 2 * i];
        b.p[i] = (const float*)d_in[3 + 2 * i];
    }

    char* ws = (char*)d_ws;
    unsigned short* pw   = (unsigned short*)ws;                       // 9*32KB packed bf16
    float*          bias = (float*)(ws + 9 * MATW * 2);               // 3*128 f32
    unsigned short* bufA = (unsigned short*)(ws + (1 << 20));         // 4096*64*128 bf16
    unsigned short* bufB = bufA + (size_t)4096 * 64 * 128;            // 4096*32*128 bf16

    // 1) pack weights + fold biases
    tree_gru_pack<<<(9 * MATW + 255) / 256, 256, 0, stream>>>(w, b, pw, bias);

    // 2) level 0: 262144 rows / 64 per block
    tree_gru_level0<<<4096, 128, 0, stream>>>(tokens, emb, pw, bias, bufA);

    // 3) tree reduction: n = 64 -> 1
    unsigned short* src = bufA;
    unsigned short* dst = bufB;
    for (int n2 = 32; n2 >= 1; n2 >>= 1) {
        int blocks = 64 * n2;  // (4096 * n2) parents / 64 per block
        int fin = (n2 == 1) ? 1 : 0;
        tree_gru_reduce<<<blocks, 128, 0, stream>>>(src, pw, bias, dst,
                                                    (float*)d_out, n2, fin);
        unsigned short* t = src; src = dst; dst = t;
    }
}
